// HeteroGNN_12343736009221
// MI455X (gfx1250) — matrix operand order repaired and checked by reading, not compiled, not run on hardware
//
#include <hip/hip_runtime.h>
#include <stdint.h>

// ---- problem constants (from reference) ----
#define NPRD 500000
#define NBRD 20000
#define NCAT 5000
#define NSHP 2000
#define NTOT (NPRD + NBRD + NCAT + NSHP)   // 527000
#define NEDG 500000
#define HDIM 64
#define ODIM 32
#define FIN  384

#define PROJ_ROWS 32                 // rows per block in k_proj (500000 % 32 == 0)
#define APF (FIN + 4)                // fp32 LDS pitch (proj): rows land on banks 4r
#define SPF (HDIM + 4)               // fp32 LDS pitch (sage): 68 dwords, same property

typedef __attribute__((ext_vector_type(16))) __bf16 v16bf;
typedef __attribute__((ext_vector_type(8)))  float  v8f;

static __device__ __forceinline__ void cvt4(v16bf& v, int base, float4 u, float s) {
    v[base + 0] = (__bf16)(u.x * s);
    v[base + 1] = (__bf16)(u.y * s);
    v[base + 2] = (__bf16)(u.z * s);
    v[base + 3] = (__bf16)(u.w * s);
}

// A fragment (16x32 bf16) from a fp32 row (global or LDS), scaled.
// lane holds row (lane&15); K elems: [k0+half*8 .. +7] and [k0+16+half*8 .. +7]
static __device__ __forceinline__ v16bf load_a_row(const float* __restrict__ rowp,
                                                   int k0, int half, float scale) {
    v16bf a;
    const float* p0 = rowp + k0 + half * 8;
    cvt4(a, 0,  *(const float4*)(p0 + 0),  scale);
    cvt4(a, 4,  *(const float4*)(p0 + 4),  scale);
    cvt4(a, 8,  *(const float4*)(p0 + 16), scale);
    cvt4(a, 12, *(const float4*)(p0 + 20), scale);
    return a;
}

// B fragment (32x16 bf16) from row-major W[outc][ldw]: B(k,n) = W[n*ldw + k].
// lane holds col (lane&15); K elems: [k0+half*8 .. +7] and [k0+16+half*8 .. +7]
static __device__ __forceinline__ v16bf load_b_row(const float* __restrict__ W,
                                                   int ldw, int col, int k0, int half) {
    v16bf b;
    const float* p = W + (size_t)col * ldw + k0 + half * 8;
    cvt4(b, 0,  *(const float4*)(p + 0),  1.0f);
    cvt4(b, 4,  *(const float4*)(p + 4),  1.0f);
    cvt4(b, 8,  *(const float4*)(p + 16), 1.0f);
    cvt4(b, 12, *(const float4*)(p + 20), 1.0f);
    return b;
}

// issue one async global->LDS 16-byte copy (GV mode, ASYNCcnt-tracked)
static __device__ __forceinline__ void async_b128(uint32_t lds_byte_addr, const float* gp) {
    asm volatile("global_load_async_to_lds_b128 %0, %1, off"
                 :: "v"(lds_byte_addr), "v"(gp) : "memory");
}
static __device__ __forceinline__ void async_wait0() {
    asm volatile("s_wait_asynccnt 0" ::: "memory");
}

// ---------------------------------------------------------------------------
// Kernel 1: prod = relu(x_product @ W_proj^T + b_proj)  -> x0[0:NPRD]
// 1 block = 32 output rows, 4 waves (16 cols each, 2 M-tiles each), K=384.
// A tile staged fp32 global->LDS via GLOBAL_LOAD_ASYNC_TO_LDS_B128,
// B fragments reused across both M-tiles.
// ---------------------------------------------------------------------------
__global__ void __launch_bounds__(128) k_proj(const float* __restrict__ X,
                                              const float* __restrict__ W,
                                              const float* __restrict__ bias,
                                              float* __restrict__ out) {
    __shared__ __align__(16) float smem[PROJ_ROWS * APF];
    const int m0   = blockIdx.x * PROJ_ROWS;
    const int tid  = threadIdx.x;
    const int wave = tid >> 5;
    const int lane = tid & 31;
    const int lrow = lane & 15;
    const int half = lane >> 4;
    const int n0   = wave * 16;

    // async stage: 32 rows x 384 f32 = 3072 x 16B chunks; 128 threads x 24 issues.
    {
        const uint32_t lds_base = (uint32_t)(uintptr_t)(void*)smem;
#pragma unroll
        for (int it = 0; it < (PROJ_ROWS * (FIN / 4)) / 128; ++it) {
            const int c    = it * 128 + tid;       // 16B-chunk id
            const int r    = c / (FIN / 4);        // row in tile
            const int col4 = (c % (FIN / 4)) * 4;  // float offset in row
            async_b128(lds_base + (uint32_t)(r * APF + col4) * 4u,
                       X + (size_t)(m0 + r) * FIN + col4);
        }
        async_wait0();
    }
    __syncthreads();

    v8f acc0 = {}, acc1 = {};
#pragma unroll
    for (int k0 = 0; k0 < FIN; k0 += 32) {
        const v16bf b  = load_b_row(W, FIN, n0 + lrow, k0, half);
        const v16bf a0 = load_a_row(&smem[lrow * APF], k0, half, 1.0f);
        acc0 = __builtin_amdgcn_wmma_f32_16x16x32_bf16(false, a0, false, b,
                                                       (short)0, acc0, false, false);
        const v16bf a1 = load_a_row(&smem[(16 + lrow) * APF], k0, half, 1.0f);
        acc1 = __builtin_amdgcn_wmma_f32_16x16x32_bf16(false, a1, false, b,
                                                       (short)0, acc1, false, false);
    }

    const float bn = bias[n0 + lrow];
#pragma unroll
    for (int v = 0; v < 8; ++v) {
        const int mA = m0 + v + half * 8;
        out[(size_t)mA * HDIM + n0 + lrow] = fmaxf(acc0[v] + bn, 0.0f);
        const int mB = m0 + 16 + v + half * 8;
        out[(size_t)mB * HDIM + n0 + lrow] = fmaxf(acc1[v] + bn, 0.0f);
    }
}

// ---------------------------------------------------------------------------
// Kernel 2: concat embeddings into x0 rows [NPRD, NTOT)
// ---------------------------------------------------------------------------
__global__ void __launch_bounds__(256) k_copy_emb(const float4* __restrict__ eb,
                                                  const float4* __restrict__ ec,
                                                  const float4* __restrict__ es,
                                                  float4* __restrict__ dst) {
    const long i   = (long)blockIdx.x * 256 + threadIdx.x;
    const long nb4 = (long)NBRD * HDIM / 4;
    const long nc4 = (long)NCAT * HDIM / 4;
    const long ns4 = (long)NSHP * HDIM / 4;
    if (i < nb4)                  dst[i] = eb[i];
    else if (i < nb4 + nc4)       dst[i] = ec[i - nb4];
    else if (i < nb4 + nc4 + ns4) dst[i] = es[i - nb4 - nc4];
}

// ---------------------------------------------------------------------------
// Kernel 3: zero agg [NTOT*HDIM] (+ cnt [NTOT] for layer 1)
// ---------------------------------------------------------------------------
template <bool ZERO_CNT>
__global__ void __launch_bounds__(256) k_zero(float4* __restrict__ agg,
                                              float4* __restrict__ cnt) {
    const long i   = (long)blockIdx.x * 256 + threadIdx.x;
    const long na4 = (long)NTOT * HDIM / 4;
    const float4 z = make_float4(0.f, 0.f, 0.f, 0.f);
    if (i < na4)                                  agg[i] = z;
    else if (ZERO_CNT && i < na4 + (long)NTOT / 4) cnt[i - na4] = z;
}

// ---------------------------------------------------------------------------
// Kernel 4: edge scatter (both directions per undirected edge pair).
// 16 lanes per edge; each lane moves 4 floats + f32 atomics into agg.
// cnt (node degree) only accumulated on the first layer.
// ---------------------------------------------------------------------------
template <bool UPDATE_CNT>
__global__ void __launch_bounds__(256) k_scatter(const int* __restrict__ pb_s, const int* __restrict__ pb_d,
                                                 const int* __restrict__ pc_s, const int* __restrict__ pc_d,
                                                 const int* __restrict__ ps_s, const int* __restrict__ ps_d,
                                                 const float* __restrict__ x,
                                                 float* __restrict__ agg,
                                                 float* __restrict__ cnt) {
    const long t  = (long)blockIdx.x * 256 + threadIdx.x;
    const long eg = t >> 4;
    if (eg >= 3L * NEDG) return;
    const int lane4 = (int)(t & 15);
    const int rel   = (int)(eg / NEDG);
    const int i     = (int)(eg - (long)rel * NEDG);

    int s, d;
    if (rel == 0)      { s = pb_s[i]; d = NPRD + pb_d[i]; }
    else if (rel == 1) { s = pc_s[i]; d = NPRD + NBRD + pc_d[i]; }
    else               { s = ps_s[i]; d = NPRD + NBRD + NCAT + ps_d[i]; }

    const float4 xs = *(const float4*)(x + (size_t)s * HDIM + lane4 * 4);
    const float4 xd = *(const float4*)(x + (size_t)d * HDIM + lane4 * 4);
    float* ad = agg + (size_t)d * HDIM + lane4 * 4;
    float* as = agg + (size_t)s * HDIM + lane4 * 4;
    atomicAdd(ad + 0, xs.x); atomicAdd(ad + 1, xs.y);
    atomicAdd(ad + 2, xs.z); atomicAdd(ad + 3, xs.w);
    atomicAdd(as + 0, xd.x); atomicAdd(as + 1, xd.y);
    atomicAdd(as + 2, xd.z); atomicAdd(as + 3, xd.w);
    if (UPDATE_CNT && lane4 == 0) { atomicAdd(cnt + d, 1.0f); atomicAdd(cnt + s, 1.0f); }
}

// ---------------------------------------------------------------------------
// Kernel 5: SAGE update: out = act( (agg/max(cnt,1)) @ Wl^T + bl + x @ Wr^T )
// 1 block = 16 rows, OUTC/16 waves (16 cols each), K=64.
// agg & x row-tiles staged fp32 global->LDS (async, ASYNCcnt); each wave then
// reads bf16 fragments from LDS; mean-normalization fused into the A convert.
// ---------------------------------------------------------------------------
template <int OUTC, bool RELU>
__global__ void __launch_bounds__(OUTC * 2) k_sage(const float* __restrict__ xin,
                                                   const float* __restrict__ agg,
                                                   const float* __restrict__ cnt,
                                                   const float* __restrict__ Wl,
                                                   const float* __restrict__ bl,
                                                   const float* __restrict__ Wr,
                                                   float* __restrict__ out) {
    __shared__ __align__(16) float sA[16 * SPF];   // agg tile (raw)
    __shared__ __align__(16) float sX[16 * SPF];   // x tile
    const int m0   = blockIdx.x * 16;
    const int lane = threadIdx.x & 31;
    const int wave = threadIdx.x >> 5;
    const int lrow = lane & 15;
    const int half = lane >> 4;
    const int n0   = wave * 16;

    // async stage: 16 rows x 64 f32 x 2 matrices = 2 x 256 chunks of 16B
    {
        const uint32_t ldsA = (uint32_t)(uintptr_t)(void*)sA;
        const uint32_t ldsX = (uint32_t)(uintptr_t)(void*)sX;
        const int nthr = OUTC * 2;
#pragma unroll
        for (int it = 0; it < 256 / nthr; ++it) {
            const int c    = it * nthr + (int)threadIdx.x;  // 0..255
            const int r    = c >> 4;                        // row in tile
            const int col4 = (c & 15) * 4;                  // float offset
            int grow = m0 + r;
            if (grow > NTOT - 1) grow = NTOT - 1;           // clamp (tail tile)
            const uint32_t off = (uint32_t)(r * SPF + col4) * 4u;
            async_b128(ldsA + off, agg + (size_t)grow * HDIM + col4);
            async_b128(ldsX + off, xin + (size_t)grow * HDIM + col4);
        }
        async_wait0();
    }
    __syncthreads();

    int row = m0 + lrow;
    if (row > NTOT - 1) row = NTOT - 1;   // clamp: keep EXEC all-ones for WMMA
    const float inv = 1.0f / fmaxf(cnt[row], 1.0f);
    const float* arow = &sA[lrow * SPF];
    const float* xrow = &sX[lrow * SPF];

    v8f acc = {};
#pragma unroll
    for (int k0 = 0; k0 < HDIM; k0 += 32) {
        const v16bf aA = load_a_row(arow, k0, half, inv);
        const v16bf bA = load_b_row(Wl, HDIM, n0 + lrow, k0, half);
        acc = __builtin_amdgcn_wmma_f32_16x16x32_bf16(false, aA, false, bA,
                                                      (short)0, acc, false, false);
        const v16bf aX = load_a_row(xrow, k0, half, 1.0f);
        const v16bf bX = load_b_row(Wr, HDIM, n0 + lrow, k0, half);
        acc = __builtin_amdgcn_wmma_f32_16x16x32_bf16(false, aX, false, bX,
                                                      (short)0, acc, false, false);
    }

    const float bn = bl[n0 + lrow];
#pragma unroll
    for (int v = 0; v < 8; ++v) {
        const int m = m0 + v + half * 8;
        if (m < NTOT) {
            float r = acc[v] + bn;
            if (RELU) r = fmaxf(r, 0.0f);
            out[(size_t)m * OUTC + n0 + lrow] = r;
        }
    }
}

// ---------------------------------------------------------------------------
extern "C" void kernel_launch(void* const* d_in, const int* in_sizes, int n_in,
                              void* d_out, int out_size, void* d_ws, size_t ws_size,
                              hipStream_t stream) {
    (void)in_sizes; (void)n_in; (void)out_size; (void)ws_size;

    const float* x_product = (const float*)d_in[0];
    const int*   pb_s = (const int*)d_in[1];
    const int*   pb_d = (const int*)d_in[2];
    const int*   pc_s = (const int*)d_in[3];
    const int*   pc_d = (const int*)d_in[4];
    const int*   ps_s = (const int*)d_in[5];
    const int*   ps_d = (const int*)d_in[6];
    const float* W_proj = (const float*)d_in[7];
    const float* b_proj = (const float*)d_in[8];
    const float* emb_b  = (const float*)d_in[9];
    const float* emb_c  = (const float*)d_in[10];
    const float* emb_s  = (const float*)d_in[11];
    const float* W1_l = (const float*)d_in[12];
    const float* b1_l = (const float*)d_in[13];
    const float* W1_r = (const float*)d_in[14];
    const float* W2_l = (const float*)d_in[15];
    const float* b2_l = (const float*)d_in[16];
    const float* W2_r = (const float*)d_in[17];

    float* ws  = (float*)d_ws;
    float* x0  = ws;                                // [NTOT, 64]
    float* x1  = ws + (size_t)NTOT * HDIM;          // [NTOT, 64]
    float* agg = ws + 2 * (size_t)NTOT * HDIM;      // [NTOT, 64]
    float* cnt = ws + 3 * (size_t)NTOT * HDIM;      // [NTOT]

    // 1) projection into x0 rows [0, NPRD)
    k_proj<<<NPRD / PROJ_ROWS, 128, 0, stream>>>(x_product, W_proj, b_proj, x0);

    // 2) embeddings into x0 rows [NPRD, NTOT)
    {
        const long total4 = (long)(NBRD + NCAT + NSHP) * HDIM / 4;
        const int  grid   = (int)((total4 + 255) / 256);
        k_copy_emb<<<grid, 256, 0, stream>>>((const float4*)emb_b, (const float4*)emb_c,
                                             (const float4*)emb_s,
                                             (float4*)(x0 + (size_t)NPRD * HDIM));
    }

    const long zall4  = (long)NTOT * HDIM / 4 + (long)NTOT / 4;
    const long zagg4  = (long)NTOT * HDIM / 4;
    const int  zgrid1 = (int)((zall4 + 255) / 256);
    const int  zgrid2 = (int)((zagg4 + 255) / 256);
    const int  sgrid  = (int)((3L * NEDG * 16 + 255) / 256);
    const int  ngrid  = (NTOT + 15) / 16;

    // ---- layer 1 (builds cnt once; cnt is reused by layer 2) ----
    k_zero<true><<<zgrid1, 256, 0, stream>>>((float4*)agg, (float4*)cnt);
    k_scatter<true><<<sgrid, 256, 0, stream>>>(pb_s, pb_d, pc_s, pc_d, ps_s, ps_d,
                                               x0, agg, cnt);
    k_sage<HDIM, true><<<ngrid, HDIM * 2, 0, stream>>>(x0, agg, cnt, W1_l, b1_l, W1_r, x1);

    // ---- layer 2 ----
    k_zero<false><<<zgrid2, 256, 0, stream>>>((float4*)agg, (float4*)cnt);
    k_scatter<false><<<sgrid, 256, 0, stream>>>(pb_s, pb_d, pc_s, pc_d, ps_s, ps_d,
                                                x1, agg, cnt);
    k_sage<ODIM, false><<<ngrid, ODIM * 2, 0, stream>>>(x1, agg, cnt, W2_l, b2_l, W2_r,
                                                        (float*)d_out);
}
